// StructureModuleLayer_86955907875217
// MI455X (gfx1250) — matrix operand order repaired and checked by reading, not compiled, not run on hardware
//
#include <hip/hip_runtime.h>
#include <hip/hip_bf16.h>
#include <math.h>

// ---------------------------------------------------------------------------
// Model constants (reference): B=1, N=512, CS=384, CZ=128, H=12, C=16,
// PQ=4, PV=8. OUT_DIM = H*(C+CZ+PV*4) = 2112.
// ---------------------------------------------------------------------------

typedef __attribute__((ext_vector_type(16))) __bf16 v16bf;
typedef __attribute__((ext_vector_type(8)))  float  v8f;
typedef __attribute__((ext_vector_type(4)))  unsigned int u32x4;
typedef __attribute__((ext_vector_type(8)))  int  i32x8;
typedef __attribute__((ext_vector_type(4)))  int  i32x4;

#define LDS_A_PITCH 36   // 32 data + 4 pad dwords (16B-aligned rows)
#define LDS_B_PITCH 68   // 64 data + 4 pad dwords

// ---------------------------------------------------------------------------
// Issue one TDM 2D tile load (rows x cols fp32, row stride ld elements) from
// global into LDS at byte offset ldsOff. Rows are padded in LDS by 4 dwords
// (pad_amount code 3) after every `cols` dwords (pad_interval code).
// D# layout per CDNA5 ISA 8.3/8.4. This toolchain's builtin takes 6 args
// (groups 0..3 + extra group + cpol).
// ---------------------------------------------------------------------------
#if __has_builtin(__builtin_amdgcn_tensor_load_to_lds)
#define HAVE_TDM 1
__device__ __forceinline__ void tdm_load_tile(const float* gsrc, unsigned ldsOff,
                                              int rows, int cols, int ld,
                                              unsigned padIntervalCode)
{
    unsigned long long ga = (unsigned long long)(uintptr_t)gsrc;
    u32x4 g0;
    g0[0] = 1u;                                        // count=1, user desc
    g0[1] = ldsOff;                                    // lds_addr
    g0[2] = (unsigned)(ga & 0xFFFFFFFFu);              // global_addr lo
    g0[3] = (unsigned)((ga >> 32) & 0x01FFFFFFu)       // global_addr hi (56:32)
            | 0x80000000u;                             // type=2 ("image")
    i32x8 g1;
    g1[0] = (int)(0x00020000u                          // data_size=2 (4 bytes)
                  | (1u << 20)                         // pad_enable
                  | (padIntervalCode << 22)            // pad_interval
                  | (3u << 25));                       // pad_amount = 4 dwords
    g1[1] = (int)(((unsigned)ld & 0xFFFFu) << 16);     // tensor_dim0[15:0]
    g1[2] = (int)((((unsigned)ld >> 16) & 0xFFFFu)     // tensor_dim0[31:16]
                  | (((unsigned)rows & 0xFFFFu) << 16)); // tensor_dim1[15:0]
    g1[3] = (int)(((unsigned)cols & 0xFFFFu) << 16);   // tile_dim0
    g1[4] = (int)((unsigned)rows & 0xFFFFu);           // tile_dim1 (tile_dim2=0)
    g1[5] = (int)(unsigned)ld;                         // tensor_dim0_stride lo
    g1[6] = 0;                                         // stride hi / dim1_stride
    g1[7] = 0;
    i32x4 z4 = {0, 0, 0, 0};
    i32x8 z8 = {0, 0, 0, 0, 0, 0, 0, 0};
    __builtin_amdgcn_tensor_load_to_lds(g0, g1, z4, z4, z8, 0);
}
#else
#define HAVE_TDM 0
#endif

// ---------------------------------------------------------------------------
// Generic batched GEMM: C = act(alpha * A@B + bias + residual)
//   A: M x K row-major (lda, per-batch element offset batchA)
//   B: K x N row-major (ldb) or, if transB, B[k][n] = Bp[n*ldb + k]
//   C: M x N row-major (ldc, per-batch offset batchC; residual shares layout)
// bf16 WMMA compute (v_wmma_f32_16x16x32_bf16), fp32 accumulate.
// Block: 256 threads = 8 waves; block tile 32(M) x 64(N); wave tile 16x16.
// Full interior tiles are fed by the Tensor Data Mover (TENSORcnt) when
// tdm!=0, or by float4 vector loads; edge tiles use branchless scalar loads
// (zero-filled), so K=12/16 zero-padding is free.
// ---------------------------------------------------------------------------
__global__ __launch_bounds__(256)
void gemm_wmma_kernel(const float* __restrict__ A, const float* __restrict__ B,
                      const float* __restrict__ bias,
                      const float* __restrict__ residual,
                      float* __restrict__ C,
                      int M, int N, int K,
                      int lda, int ldb, int ldc,
                      long batchA, long batchB, long batchC,
                      int transB, int act, float alpha, int tdm)
{
    __shared__ float sA[32][LDS_A_PITCH];
    __shared__ float sB[32][LDS_B_PITCH];

    const int tid  = threadIdx.x;
    const int lane = tid & 31;
    const int wave = tid >> 5;
    const int wm   = wave >> 2;   // 0..1  (M sub-tile)
    const int wn   = wave & 3;    // 0..3  (N sub-tile)
    const int ll   = lane & 15;
    const int lh   = lane >> 4;

    const int mBase = blockIdx.x * 32;
    const int nBase = blockIdx.y * 64;
    const long ba   = (long)blockIdx.z;

    const float* Ap = A + ba * batchA;
    const float* Bp = B + ba * batchB;
    float*       Cp = C + ba * batchC;
    const float* Rp = residual ? (residual + ba * batchC) : nullptr;

    v8f acc = {0.f, 0.f, 0.f, 0.f, 0.f, 0.f, 0.f, 0.f};

    for (int k0 = 0; k0 < K; k0 += 32) {
        __syncthreads();   // previous tile fully consumed before overwrite

        const bool aFull = (mBase + 32 <= M) && (k0 + 32 <= K);
        const bool bFull = (nBase + 64 <= N) && (k0 + 32 <= K) && !transB;
        bool tdmA = false, tdmB = false;
#if HAVE_TDM
        tdmA = tdm && aFull;
        tdmB = tdm && bFull;
        if (wave == 0 && (tdmA || tdmB)) {
            if (tdmA)
                tdm_load_tile(Ap + (long)mBase * lda + k0,
                              (unsigned)(uintptr_t)&sA[0][0],
                              32, 32, lda, 4u /* pad every 32 dwords */);
            if (tdmB)
                tdm_load_tile(Bp + (long)k0 * ldb + nBase,
                              (unsigned)(uintptr_t)&sB[0][0],
                              32, 64, ldb, 5u /* pad every 64 dwords */);
        }
#endif
        // ---- stage A tile: 32x32 floats ----
        if (!tdmA) {
            if (aFull) {
                int r = tid >> 3, c4 = (tid & 7) * 4;      // 256 float4
                const float4 v4 =
                    *(const float4*)(Ap + (long)(mBase + r) * lda + k0 + c4);
                *(float4*)&sA[r][c4] = v4;
            } else {
#pragma unroll
                for (int rep = 0; rep < 4; ++rep) {
                    int idx = tid + rep * 256;
                    int r = idx >> 5, c = idx & 31;
                    int m = mBase + r, k = k0 + c;
                    bool ok = (m < M) && (k < K);
                    float val = Ap[ok ? ((long)m * lda + k) : 0];
                    sA[r][c] = ok ? val : 0.f;
                }
            }
        }
        // ---- stage B tile: 32x64 floats ----
        if (!tdmB) {
            if (bFull) {
#pragma unroll
                for (int rep = 0; rep < 2; ++rep) {
                    int f4 = tid + rep * 256;              // 512 float4
                    int r = f4 >> 4, c4 = (f4 & 15) * 4;
                    const float4 v4 =
                        *(const float4*)(Bp + (long)(k0 + r) * ldb + nBase + c4);
                    *(float4*)&sB[r][c4] = v4;
                }
            } else {
#pragma unroll
                for (int rep = 0; rep < 8; ++rep) {
                    int idx = tid + rep * 256;
                    int r = idx >> 6, c = idx & 63;
                    int k = k0 + r, n = nBase + c;
                    bool ok = (k < K) && (n < N);
                    long bidx = transB ? ((long)n * ldb + k) : ((long)k * ldb + n);
                    float val = Bp[ok ? bidx : 0];
                    sB[r][c] = ok ? val : 0.f;
                }
            }
        }
        // prefetch next K-slab toward L2 while this one is consumed
        if (k0 + 32 < K) {
            int pm = mBase + (tid & 31);
            if (pm < M) __builtin_prefetch(&Ap[(long)pm * lda + k0 + 32]);
            int pn = nBase + (tid & 63);
            if (pn < N) {
                if (transB) __builtin_prefetch(&Bp[(long)pn * ldb + k0 + 32]);
                else        __builtin_prefetch(&Bp[(long)(k0 + 32) * ldb + pn]);
            }
        }
#if HAVE_TDM
        if (wave == 0 && (tdmA || tdmB)) __builtin_amdgcn_s_wait_tensorcnt(0);
#endif
        __syncthreads();

        // ---- fragments per ISA 7.12.2 16-bit layouts ----
        // A 16x32: lanes0-15 M=lane, K in {kb..kb+7, kb+16..kb+23}, kb=lh*8
        v16bf af;
        {
            int arow = wm * 16 + ll;
            int kb = lh * 8;
#pragma unroll
            for (int u = 0; u < 4; ++u) {
                af[2 * u]         = (__bf16)sA[arow][kb + 2 * u];
                af[2 * u + 1]     = (__bf16)sA[arow][kb + 2 * u + 1];
                af[8 + 2 * u]     = (__bf16)sA[arow][kb + 16 + 2 * u];
                af[8 + 2 * u + 1] = (__bf16)sA[arow][kb + 16 + 2 * u + 1];
            }
        }
        // B 32x16: N=ll, K in {kb..kb+7, kb+16..kb+23}, kb=lh*8
        v16bf bfrag;
        {
            int bcol = wn * 16 + ll;
            int kb = lh * 8;
#pragma unroll
            for (int u = 0; u < 4; ++u) {
                bfrag[2 * u]         = (__bf16)sB[kb + 2 * u][bcol];
                bfrag[2 * u + 1]     = (__bf16)sB[kb + 2 * u + 1][bcol];
                bfrag[8 + 2 * u]     = (__bf16)sB[kb + 16 + 2 * u][bcol];
                bfrag[8 + 2 * u + 1] = (__bf16)sB[kb + 16 + 2 * u + 1][bcol];
            }
        }
        acc = __builtin_amdgcn_wmma_f32_16x16x32_bf16(
            false, af, false, bfrag, (short)0, acc, false, false);
    }

    // ---- epilogue: C layout, VGPR r -> M = wm*16 + lh*8 + r, N = ll ----
    int n = nBase + wn * 16 + ll;
    if (n < N) {
        float bv = bias ? bias[n] : 0.f;
#pragma unroll
        for (int r = 0; r < 8; ++r) {
            int m = mBase + wm * 16 + lh * 8 + r;
            if (m < M) {
                float val = alpha * acc[r] + bv;
                if (Rp) val += Rp[(long)m * ldc + n];
                if (act) val = fmaxf(val, 0.f);
                Cp[(long)m * ldc + n] = val;
            }
        }
    }
}

// ---------------------------------------------------------------------------
// Apply rigid transforms to q/k/v points and compute per-(i,h) point norms.
// qp,kp: [N][H*PQ*3]=144  vp: [N][H*PV*3]=288  rot: [N][9]  trans: [N][3]
// ---------------------------------------------------------------------------
__global__ void point_transform_kernel(const float* __restrict__ qp,
                                       const float* __restrict__ kp,
                                       const float* __restrict__ vp,
                                       const float* __restrict__ rot,
                                       const float* __restrict__ trans,
                                       float* __restrict__ qpg,
                                       float* __restrict__ kpg,
                                       float* __restrict__ vpg,
                                       float* __restrict__ qn,
                                       float* __restrict__ kn)
{
    int n = blockIdx.x;
    int t = threadIdx.x;                // 128 threads
    float R[9], T[3];
#pragma unroll
    for (int j = 0; j < 9; ++j) R[j] = rot[n * 9 + j];
#pragma unroll
    for (int j = 0; j < 3; ++j) T[j] = trans[n * 3 + j];

    for (int idx = t; idx < 48; idx += blockDim.x) {   // H*PQ q and k points
        const float* s = qp + (long)n * 144 + idx * 3;
        float x = s[0], y = s[1], z = s[2];
        float* d = qpg + (long)n * 144 + idx * 3;
        d[0] = R[0] * x + R[1] * y + R[2] * z + T[0];
        d[1] = R[3] * x + R[4] * y + R[5] * z + T[1];
        d[2] = R[6] * x + R[7] * y + R[8] * z + T[2];
        s = kp + (long)n * 144 + idx * 3;
        x = s[0]; y = s[1]; z = s[2];
        d = kpg + (long)n * 144 + idx * 3;
        d[0] = R[0] * x + R[1] * y + R[2] * z + T[0];
        d[1] = R[3] * x + R[4] * y + R[5] * z + T[1];
        d[2] = R[6] * x + R[7] * y + R[8] * z + T[2];
    }
    for (int idx = t; idx < 96; idx += blockDim.x) {   // H*PV v points
        const float* s = vp + (long)n * 288 + idx * 3;
        float x = s[0], y = s[1], z = s[2];
        float* d = vpg + (long)n * 288 + idx * 3;
        d[0] = R[0] * x + R[1] * y + R[2] * z + T[0];
        d[1] = R[3] * x + R[4] * y + R[5] * z + T[1];
        d[2] = R[6] * x + R[7] * y + R[8] * z + T[2];
    }
    __threadfence_block();
    __syncthreads();
    for (int h = t; h < 12; h += blockDim.x) {
        float sq = 0.f, sk = 0.f;
        for (int pc = 0; pc < 12; ++pc) {
            float a = qpg[(long)n * 144 + h * 12 + pc]; sq += a * a;
            float b = kpg[(long)n * 144 + h * 12 + pc]; sk += b * b;
        }
        qn[n * 12 + h] = sq;
        kn[n * 12 + h] = sk;
    }
}

// ---------------------------------------------------------------------------
// logits = wL*(qk + bias - gamma_h*(wC/2)*d2); softmax over j (row of 512).
// qk,cross,a_out: [H][N][N]; biasb: [N][N][H]; one block per (i,h).
// ---------------------------------------------------------------------------
__global__ void logits_softmax_kernel(const float* __restrict__ qk,
                                      const float* __restrict__ cross,
                                      const float* __restrict__ biasb,
                                      const float* __restrict__ qn,
                                      const float* __restrict__ kn,
                                      const float* __restrict__ head_w,
                                      float* __restrict__ a_out)
{
    const int i = blockIdx.x;
    const int h = blockIdx.y;
    const int t = threadIdx.x;          // 256 threads, 2 columns each
    const float wL = 0.57735026918962576f;             // sqrt(1/3)
    const float wC_half = 0.5f * 0.23570226039551587f; // 0.5*sqrt(2/(9*PQ))
    const float gamma = log1pf(expf(head_w[h]));
    const float gscale = gamma * wC_half;
    const float qni = qn[i * 12 + h];
    const long base = ((long)h * 512 + i) * 512;

    int j0 = t, j1 = t + 256;
    float d2 = qni + kn[j0 * 12 + h] - 2.f * cross[base + j0];
    float l0 = wL * (qk[base + j0] + biasb[((long)i * 512 + j0) * 12 + h] - gscale * d2);
    d2 = qni + kn[j1 * 12 + h] - 2.f * cross[base + j1];
    float l1 = wL * (qk[base + j1] + biasb[((long)i * 512 + j1) * 12 + h] - gscale * d2);

    __shared__ float red[256];
    red[t] = fmaxf(l0, l1);
    __syncthreads();
    for (int s = 128; s > 0; s >>= 1) {
        if (t < s) red[t] = fmaxf(red[t], red[t + s]);
        __syncthreads();
    }
    float mx = red[0];
    __syncthreads();
    float e0 = expf(l0 - mx), e1 = expf(l1 - mx);
    red[t] = e0 + e1;
    __syncthreads();
    for (int s = 128; s > 0; s >>= 1) {
        if (t < s) red[t] += red[t + s];
        __syncthreads();
    }
    float inv = 1.f / red[0];
    a_out[base + j0] = e0 * inv;
    a_out[base + j1] = e1 * inv;
}

// ---------------------------------------------------------------------------
// Rotate o_pt back to local frame, compute norms, assemble cat row:
// [ o(192) | o_pt(288) | o_norm(96) | o_pair(1536) ]  (2112 per residue)
// ---------------------------------------------------------------------------
__global__ void build_cat_kernel(const float* __restrict__ o,
                                 const float* __restrict__ optg,
                                 const float* __restrict__ opair,
                                 const float* __restrict__ rot,
                                 const float* __restrict__ trans,
                                 float* __restrict__ cat)
{
    int n = blockIdx.x;
    int t = threadIdx.x;                // 128 threads
    float* crow = cat + (long)n * 2112;
    const float* R = rot + n * 9;
    const float* T = trans + n * 3;

    if (t < 96) {                       // (h,p): o_pt = R^T * (o_pt_g - t)
        const float* v = optg + (long)n * 288 + t * 3;
        float dx = v[0] - T[0], dy = v[1] - T[1], dz = v[2] - T[2];
        float px = R[0] * dx + R[3] * dy + R[6] * dz;
        float py = R[1] * dx + R[4] * dy + R[7] * dz;
        float pz = R[2] * dx + R[5] * dy + R[8] * dz;
        crow[192 + t * 3 + 0] = px;
        crow[192 + t * 3 + 1] = py;
        crow[192 + t * 3 + 2] = pz;
        crow[480 + t] = sqrtf(px * px + py * py + pz * pz + 1e-8f);
    }
    for (int c = t; c < 192; c += blockDim.x)
        crow[c] = o[(long)n * 192 + c];
    for (int c = t; c < 1536; c += blockDim.x)
        crow[576 + c] = opair[(long)n * 1536 + c];
}

// ---------------------------------------------------------------------------
// Row layernorm (D=384): one block per row.
// ---------------------------------------------------------------------------
__global__ void layernorm_kernel(const float* __restrict__ x,
                                 const float* __restrict__ g,
                                 const float* __restrict__ b,
                                 float* __restrict__ y, int D)
{
    int n = blockIdx.x;
    int t = threadIdx.x;                // 128 threads
    __shared__ float red[128];
    const float* xr = x + (long)n * D;
    float s = 0.f;
    for (int c = t; c < D; c += blockDim.x) s += xr[c];
    red[t] = s;
    __syncthreads();
    for (int o = 64; o > 0; o >>= 1) {
        if (t < o) red[t] += red[t + o];
        __syncthreads();
    }
    float mean = red[0] / D;
    __syncthreads();
    float v = 0.f;
    for (int c = t; c < D; c += blockDim.x) { float d = xr[c] - mean; v += d * d; }
    red[t] = v;
    __syncthreads();
    for (int o = 64; o > 0; o >>= 1) {
        if (t < o) red[t] += red[t + o];
        __syncthreads();
    }
    float inv = rsqrtf(red[0] / D + 1e-5f);
    float* yr = y + (long)n * D;
    for (int c = t; c < D; c += blockDim.x)
        yr[c] = (xr[c] - mean) * inv * g[c] + b[c];
}

// ---------------------------------------------------------------------------
// Backbone update: upd = x @ w_bb + b_bb; quat->R_u; R_new = R@R_u;
// t_new = R@t_u + trans. One thread per residue.
// ---------------------------------------------------------------------------
__global__ void backbone_kernel(const float* __restrict__ x,
                                const float* __restrict__ w_bb,
                                const float* __restrict__ b_bb,
                                const float* __restrict__ rot,
                                const float* __restrict__ trans,
                                float* __restrict__ Rout,
                                float* __restrict__ Tout)
{
    int n = blockIdx.x * blockDim.x + threadIdx.x;
    if (n >= 512) return;
    float u[6];
#pragma unroll
    for (int c = 0; c < 6; ++c) u[c] = b_bb[c];
    const float* xr = x + (long)n * 384;
    for (int k = 0; k < 384; ++k) {
        float xv = xr[k];
        const float* wr = w_bb + k * 6;
#pragma unroll
        for (int c = 0; c < 6; ++c) u[c] += xv * wr[c];
    }
    float b = u[0], c = u[1], d = u[2];
    float inv = rsqrtf(1.f + b * b + c * c + d * d);
    float a = inv; b *= inv; c *= inv; d *= inv;
    float Ru[9] = {
        a * a + b * b - c * c - d * d, 2.f * (b * c - a * d), 2.f * (b * d + a * c),
        2.f * (b * c + a * d), a * a - b * b + c * c - d * d, 2.f * (c * d - a * b),
        2.f * (b * d - a * c), 2.f * (c * d + a * b), a * a - b * b - c * c + d * d };
    const float* R = rot + n * 9;
    float* Ro = Rout + n * 9;
#pragma unroll
    for (int i = 0; i < 3; ++i)
#pragma unroll
        for (int j = 0; j < 3; ++j) {
            float s = 0.f;
#pragma unroll
            for (int kk = 0; kk < 3; ++kk) s += R[i * 3 + kk] * Ru[kk * 3 + j];
            Ro[i * 3 + j] = s;
        }
    float* To = Tout + n * 3;
    To[0] = R[0] * u[3] + R[1] * u[4] + R[2] * u[5] + trans[n * 3 + 0];
    To[1] = R[3] * u[3] + R[4] * u[4] + R[5] * u[5] + trans[n * 3 + 1];
    To[2] = R[6] * u[3] + R[7] * u[4] + R[8] * u[5] + trans[n * 3 + 2];
}

// ---------------------------------------------------------------------------
// Host-side orchestration. Input order (setup_inputs dict, params flattened):
//  0 single 1 pair 2 rotations 3 translations 4 w_q 5 b_q 6 w_k 7 b_k
//  8 w_v 9 b_v 10 w_qp 11 b_qp 12 w_kp 13 b_kp 14 w_vp 15 b_vp 16 w_b 17 b_b
//  18 head_weights 19 w_out 20 b_out 21 ln1_g 22 ln1_b 23 w1 24 b1 25 w2
//  26 b2 27 w3 28 b3 29 ln2_g 30 ln2_b 31 w_bb 32 b_bb
// Output: (x[512*384], R_new[512*9], t_new[512*3]) concatenated.
// ---------------------------------------------------------------------------
extern "C" void kernel_launch(void* const* d_in, const int* in_sizes, int n_in,
                              void* d_out, int out_size, void* d_ws, size_t ws_size,
                              hipStream_t stream)
{
    const float* single = (const float*)d_in[0];
    const float* pair   = (const float*)d_in[1];
    const float* rot    = (const float*)d_in[2];
    const float* trans  = (const float*)d_in[3];
    const float* w_q  = (const float*)d_in[4];  const float* b_q  = (const float*)d_in[5];
    const float* w_k  = (const float*)d_in[6];  const float* b_k  = (const float*)d_in[7];
    const float* w_v  = (const float*)d_in[8];  const float* b_v  = (const float*)d_in[9];
    const float* w_qp = (const float*)d_in[10]; const float* b_qp = (const float*)d_in[11];
    const float* w_kp = (const float*)d_in[12]; const float* b_kp = (const float*)d_in[13];
    const float* w_vp = (const float*)d_in[14]; const float* b_vp = (const float*)d_in[15];
    const float* w_b  = (const float*)d_in[16]; const float* b_b  = (const float*)d_in[17];
    const float* head_w = (const float*)d_in[18];
    const float* w_out = (const float*)d_in[19]; const float* b_out = (const float*)d_in[20];
    const float* ln1_g = (const float*)d_in[21]; const float* ln1_b = (const float*)d_in[22];
    const float* w1 = (const float*)d_in[23]; const float* b1 = (const float*)d_in[24];
    const float* w2 = (const float*)d_in[25]; const float* b2 = (const float*)d_in[26];
    const float* w3 = (const float*)d_in[27]; const float* b3 = (const float*)d_in[28];
    const float* ln2_g = (const float*)d_in[29]; const float* ln2_b = (const float*)d_in[30];
    const float* w_bb = (const float*)d_in[31]; const float* b_bb = (const float*)d_in[32];

    float* out = (float*)d_out;
    float* ws  = (float*)d_ws;
    size_t off = 0;
    auto alloc = [&](size_t nelem) { float* p = ws + off; off += nelem; return p; };

    float* qb   = alloc(512 * 192);
    float* kb   = alloc(512 * 192);
    float* vb   = alloc(512 * 192);
    float* qp   = alloc(512 * 144);
    float* kp   = alloc(512 * 144);
    float* vp   = alloc(512 * 288);
    float* qpg  = alloc(512 * 144);
    float* kpg  = alloc(512 * 144);
    float* vpg  = alloc(512 * 288);
    float* qn   = alloc(512 * 12);
    float* kn   = alloc(512 * 12);
    float* qkB  = alloc(12L * 512 * 512);
    float* crsB = alloc(12L * 512 * 512);
    float* bsB  = alloc(512L * 512 * 12);
    float* att  = alloc(12L * 512 * 512);
    float* oB   = alloc(512 * 192);
    float* optg = alloc(512 * 288);
    float* opr  = alloc(512 * 1536);
    float* cat  = alloc(512 * 2112);
    float* x0   = alloc(512 * 384);
    float* x1   = alloc(512 * 384);
    float* h1   = alloc(512 * 384);
    float* h2   = alloc(512 * 384);
    float* x2   = alloc(512 * 384);

    auto gemm = [&](const float* A, const float* B, const float* bias,
                    const float* res, float* C, int M, int N, int K,
                    int lda, int ldb, int ldc, long bA, long bB, long bC,
                    int batch, int transB, int act, float alpha) {
        int tdm = (!transB && (K % 32 == 0)) ? 1 : 0;  // TDM path eligibility
        dim3 g((M + 31) / 32, (N + 63) / 64, batch);
        gemm_wmma_kernel<<<g, 256, 0, stream>>>(A, B, bias, res, C, M, N, K,
                                                lda, ldb, ldc, bA, bB, bC,
                                                transB, act, alpha, tdm);
    };

    // --- projections: single @ W + b ---
    gemm(single, w_q,  b_q,  nullptr, qb, 512, 192, 384, 384, 192, 192, 0, 0, 0, 1, 0, 0, 1.f);
    gemm(single, w_k,  b_k,  nullptr, kb, 512, 192, 384, 384, 192, 192, 0, 0, 0, 1, 0, 0, 1.f);
    gemm(single, w_v,  b_v,  nullptr, vb, 512, 192, 384, 384, 192, 192, 0, 0, 0, 1, 0, 0, 1.f);
    gemm(single, w_qp, b_qp, nullptr, qp, 512, 144, 384, 384, 144, 144, 0, 0, 0, 1, 0, 0, 1.f);
    gemm(single, w_kp, b_kp, nullptr, kp, 512, 144, 384, 384, 144, 144, 0, 0, 0, 1, 0, 0, 1.f);
    gemm(single, w_vp, b_vp, nullptr, vp, 512, 288, 384, 384, 288, 288, 0, 0, 0, 1, 0, 0, 1.f);

    point_transform_kernel<<<512, 128, 0, stream>>>(qp, kp, vp, rot, trans,
                                                    qpg, kpg, vpg, qn, kn);

    // --- qk per head (transB view of k), alpha = 1/sqrt(C) = 0.25 ---
    gemm(qb, kb, nullptr, nullptr, qkB, 512, 512, 16, 192, 192, 512,
         16, 16, 512L * 512, 12, 1, 0, 0.25f);
    // --- point cross term per head, K = PQ*3 = 12 (zero-padded in tile) ---
    gemm(qpg, kpg, nullptr, nullptr, crsB, 512, 512, 12, 144, 144, 512,
         12, 12, 512L * 512, 12, 1, 0, 1.f);
    // --- pair bias: (N*N) x CZ x H, first pass over pair (134 MB) ---
    gemm(pair, w_b, b_b, nullptr, bsB, 512 * 512, 12, 128, 128, 12, 12,
         0, 0, 0, 1, 0, 0, 1.f);

    logits_softmax_kernel<<<dim3(512, 12), 256, 0, stream>>>(qkB, crsB, bsB,
                                                             qn, kn, head_w, att);

    // --- o = a @ v per head ---
    gemm(att, vb, nullptr, nullptr, oB, 512, 16, 512, 512, 192, 192,
         512L * 512, 16, 16, 12, 0, 0, 1.f);
    // --- o_pt_g = a @ vp_g per head (N = PV*3 = 24) ---
    gemm(att, vpg, nullptr, nullptr, optg, 512, 24, 512, 512, 288, 288,
         512L * 512, 24, 24, 12, 0, 0, 1.f);
    // --- o_pair = a^T @ pair per residue i (second pass over pair -> L2) ---
    gemm(att, pair, nullptr, nullptr, opr, 12, 128, 512, 512 * 512, 128, 128,
         512, 512L * 128, 12L * 128, 512, 0, 0, 1.f);

    build_cat_kernel<<<512, 128, 0, stream>>>(oB, optg, opr, rot, trans, cat);

    // --- ipa_out = cat @ w_out + b_out, fused residual with `single` ---
    gemm(cat, w_out, b_out, single, x0, 512, 384, 2112, 2112, 384, 384,
         0, 0, 0, 1, 0, 0, 1.f);
    layernorm_kernel<<<512, 128, 0, stream>>>(x0, ln1_g, ln1_b, x1, 384);

    // --- transition ---
    gemm(x1, w1, b1, nullptr, h1, 512, 384, 384, 384, 384, 384, 0, 0, 0, 1, 0, 1, 1.f);
    gemm(h1, w2, b2, nullptr, h2, 512, 384, 384, 384, 384, 384, 0, 0, 0, 1, 0, 1, 1.f);
    gemm(h2, w3, b3, x1, x2, 512, 384, 384, 384, 384, 384, 0, 0, 0, 1, 0, 0, 1.f);
    layernorm_kernel<<<512, 128, 0, stream>>>(x2, ln2_g, ln2_b, out, 384);

    // --- backbone update, writes R_new and t_new after x in d_out ---
    backbone_kernel<<<8, 64, 0, stream>>>(out, w_bb, b_bb, rot, trans,
                                          out + 512 * 384,
                                          out + 512 * 384 + 512 * 9);
}
